// TrendPredict_27015344292119
// MI455X (gfx1250) — matrix operand order repaired and checked by reading, not compiled, not run on hardware
//
#include <hip/hip_runtime.h>
#include <hip/hip_bf16.h>

// ---------------- problem constants ----------------
#define BB     64
#define KC     32
#define SEQ    2048
#define PRED   1990
#define PATCHW 16
#define DM     1024
#define TOTL   4038
#define NPAT   253            // (4038+10)/16
#define BN     (BB*NPAT)      // 16192  rows of the activation GEMMs
#define BKR    (BB*KC)        // 2048   rows of the trend GEMM
#define TROW   4048           // padded trend length

typedef __attribute__((ext_vector_type(16))) __bf16 v16bf;
typedef __attribute__((ext_vector_type(8)))  __bf16 v8bf;
typedef __attribute__((ext_vector_type(8)))  float  v8f;

__device__ inline v16bf mk16(v8bf lo, v8bf hi) {
  v16bf r;
#pragma unroll
  for (int i = 0; i < 8; ++i) { r[i] = lo[i]; r[i + 8] = hi[i]; }
  return r;
}

// low 32 bits of a flat LDS address = wave-relative LDS byte offset
__device__ inline unsigned lds_off(const void* p) {
  return (unsigned)(size_t)p;
}

// ---------------------------------------------------------------------------
// Generic bf16 WMMA GEMM:  C[M,Nc] = act( A[M,Kd] * B[Nc,Kd]^T + bias )
//   - block tile 64x256, K step 32, 256 threads = 8 waves (2x4),
//     wave tile 32x64 = 2x4 v_wmma_f32_16x16x32_bf16
//   - async memory->LDS DMA (ASYNCcnt), double-buffered, 1 barrier per K-step
//   - epi: 0 = identity, 1 = silu, 2 = sigmoid
//   - Cf (f32, ldc) and/or Cb (bf16, ldcb) outputs, either may be null
// ---------------------------------------------------------------------------
__global__ __launch_bounds__(256) void gemm_wmma(
    const __bf16* __restrict__ A, int lda,
    const __bf16* __restrict__ Bw, int ldb,
    const float* __restrict__ bias,
    float* __restrict__ Cf, int ldc,
    __bf16* __restrict__ Cb, int ldcb,
    int M, int Nc, int Kd, int epi)
{
  __shared__ __bf16 As[2][64][40];     // +8 pad: conflict-free b128 frag loads
  __shared__ __bf16 Bs[2][256][40];    // total 50 KB

  const int tid  = threadIdx.x;
  const int lane = tid & 31;
  const int wave = tid >> 5;
  const int laneHalf = lane >> 4;
  const int lane15   = lane & 15;
  const int waveRow  = wave >> 2;        // 0..1  -> 32-row strip
  const int waveCol  = wave & 3;         // 0..3  -> 64-col strip
  const int rowBase  = blockIdx.x * 64;
  const int colBase  = blockIdx.y * 256;

  v8f zero = {};
  v8f acc[2][4];
#pragma unroll
  for (int i = 0; i < 2; ++i)
#pragma unroll
    for (int j = 0; j < 4; ++j) acc[i][j] = zero;

  const int arow = tid >> 2, achk = (tid & 3) * 8;   // A: 64 rows x 32 k (16B/thread)
  const int bcolT = tid;                             // B: 256 cols, 64B/thread
  // Clamp B row: garbage in column n only affects output column n, which the
  // epilogue masks. Keeps async-issue count uniform across the block (no EXEC
  // divergence around the WMMA region).
  const int bcg = (colBase + bcolT < Nc) ? (colBase + bcolT) : (Nc - 1);

  const __bf16* gA = A  + (size_t)(rowBase + arow) * lda + achk;
  const __bf16* gB = Bw + (size_t)bcg * ldb;

  // async memory->LDS stage of one 64x32 A tile + 256x32 B tile
  // (5 async instructions per thread, uniform across the block)
  auto stage = [&](int k0, int b) {
    unsigned la = lds_off(&As[b][arow][achk]);
    unsigned lb = lds_off(&Bs[b][bcolT][0]);
    unsigned long long ga = (unsigned long long)(size_t)(gA + k0);
    unsigned long long gb = (unsigned long long)(size_t)(gB + k0);
    asm volatile("global_load_async_to_lds_b128 %0, %1, off"
                 :: "v"(la), "v"(ga) : "memory");
    // INST_OFFSET applies to both LDS and global address (ISA 15.18.3)
    asm volatile("global_load_async_to_lds_b128 %0, %1, off"
                 :: "v"(lb), "v"(gb) : "memory");
    asm volatile("global_load_async_to_lds_b128 %0, %1, off offset:16"
                 :: "v"(lb), "v"(gb) : "memory");
    asm volatile("global_load_async_to_lds_b128 %0, %1, off offset:32"
                 :: "v"(lb), "v"(gb) : "memory");
    asm volatile("global_load_async_to_lds_b128 %0, %1, off offset:48"
                 :: "v"(lb), "v"(gb) : "memory");
  };

  stage(0, 0);
  int buf = 0;

  for (int k0 = 0; k0 < Kd; k0 += 32) {
    // my stage(k0) ops are the only outstanding async ops -> wait to 0,
    // then barrier makes every wave's DMA-written tile visible.
    asm volatile("s_wait_asynccnt 0x0" ::: "memory");
    __syncthreads();
    // kick off next tile's DMA into the other buffer; it overlaps the WMMAs
    // below. The barrier above guarantees no wave still reads that buffer.
    if (k0 + 32 < Kd) stage(k0 + 32, buf ^ 1);

    v16bf af[2], bfr[4];
#pragma unroll
    for (int tm = 0; tm < 2; ++tm) {
      // A frag: lane m holds row m; lane-half K order {0..7,16..23}/{8..15,24..31}
      const __bf16* ap = &As[buf][waveRow * 32 + tm * 16 + lane15][0];
      af[tm] = mk16(*(const v8bf*)(ap + laneHalf * 8),
                    *(const v8bf*)(ap + laneHalf * 8 + 16));
    }
#pragma unroll
    for (int tn = 0; tn < 4; ++tn) {
      // B frag: lane n holds column n; lane-half K order {0..7,16..23}/{8..15,24..31}
      const __bf16* bp = &Bs[buf][waveCol * 64 + tn * 16 + lane15][0];
      bfr[tn] = mk16(*(const v8bf*)(bp + laneHalf * 8),
                     *(const v8bf*)(bp + laneHalf * 8 + 16));
    }
#pragma unroll
    for (int tm = 0; tm < 2; ++tm)
#pragma unroll
      for (int tn = 0; tn < 4; ++tn)
        acc[tm][tn] = __builtin_amdgcn_wmma_f32_16x16x32_bf16(
            false, af[tm], false, bfr[tn], (short)0, acc[tm][tn], false, false);

    buf ^= 1;
  }

  // epilogue: lane<16 -> M=r, lane>=16 -> M=r+8
#pragma unroll
  for (int tm = 0; tm < 2; ++tm)
#pragma unroll
    for (int tn = 0; tn < 4; ++tn) {
      int col = colBase + waveCol * 64 + tn * 16 + lane15;
      if (col >= Nc) continue;
      float bv = bias ? bias[col] : 0.0f;
#pragma unroll
      for (int r = 0; r < 8; ++r) {
        int row = rowBase + waveRow * 32 + tm * 16 + laneHalf * 8 + r;
        float v = acc[tm][tn][r] + bv;
        if (epi == 1)      v = v / (1.0f + __expf(-v));    // silu
        else if (epi == 2) v = 1.0f / (1.0f + __expf(-v)); // sigmoid
        if (Cf) Cf[(size_t)row * ldc + col] = v;
        if (Cb) Cb[(size_t)row * ldcb + col] = (__bf16)v;
      }
    }
}

// ---------------- elementwise / layout kernels ----------------
__global__ void k_conv_bf16(const float* __restrict__ s, __bf16* __restrict__ d, size_t n) {
  size_t i = (size_t)blockIdx.x * blockDim.x + threadIdx.x;
  if (i < n) d[i] = (__bf16)s[i];
}

__global__ void k_copy_x(const float* __restrict__ x, float* __restrict__ trend) {
  size_t i = (size_t)blockIdx.x * blockDim.x + threadIdx.x;
  if (i < (size_t)BKR * SEQ) {
    size_t row = i / SEQ, col = i % SEQ;
    trend[row * TROW + col] = x[i];
  }
}

__global__ void k_padfill(float* __restrict__ trend) {
  int i = blockIdx.x * blockDim.x + threadIdx.x;
  if (i < BKR * 10) {
    int row = i / 10, j = i % 10;
    trend[(size_t)row * TROW + TOTL + j] = trend[(size_t)row * TROW + TOTL - 1];
  }
}

// tpb[(b*NPAT+n)*512 + c*16+p] = trend[(b*KC+c)*TROW + n*16+p]   (bf16 im2col)
__global__ void k_patchify(const float* __restrict__ trend, __bf16* __restrict__ tpb) {
  size_t i = (size_t)blockIdx.x * blockDim.x + threadIdx.x;
  if (i < (size_t)BN * 512) {
    int p = i & 15;
    int c = (i >> 4) & 31;
    int row = (int)(i >> 9);
    int b = row / NPAT, n = row % NPAT;
    tpb[i] = (__bf16)trend[(size_t)(b * KC + c) * TROW + n * PATCHW + p];
  }
}

// shifted-half concat + exact gelu, output bf16
__global__ void k_shiftgelu(const float* __restrict__ u, __bf16* __restrict__ ub) {
  size_t i = (size_t)blockIdx.x * blockDim.x + threadIdx.x;
  if (i < (size_t)BN * DM) {
    int d = i & (DM - 1);
    int row = (int)(i >> 10);
    int n = row % NPAT;
    float v;
    if (d < DM / 2) v = (n > 0) ? u[i - DM] : 0.0f;
    else            v = u[i];
    float g = 0.5f * v * (1.0f + erff(v * 0.70710678118654752f));
    ub[i] = (__bf16)g;
  }
}

// h_n = f_n * h_{n-1} + (1-f_n) * i_n   over n=0..252, per (b,d)
__global__ void k_recur(const float* __restrict__ fb, const float* __restrict__ ib,
                        float* __restrict__ hb) {
  int t = blockIdx.x * blockDim.x + threadIdx.x;
  if (t < BB * DM) {
    int b = t >> 10, d = t & (DM - 1);
    size_t base = (size_t)b * NPAT * DM + d;
    float h = 0.0f;
    for (int n = 0; n < NPAT; ++n) {
      size_t idx = base + (size_t)n * DM;
      float fv = fb[idx], iv = ib[idx];
      h = fv * h + (1.0f - fv) * iv;
      hb[idx] = h;
    }
  }
}

// layernorm(h) * gate -> bf16
__global__ __launch_bounds__(256) void k_lngate(
    const float* __restrict__ h, const float* __restrict__ g,
    const float* __restrict__ gamma, const float* __restrict__ beta,
    __bf16* __restrict__ outb) {
  __shared__ float rs_[256], rq_[256];
  int row = blockIdx.x, tid = threadIdx.x;
  const float* hr = h + (size_t)row * DM;
  const float* gr = g + (size_t)row * DM;
  float s = 0.f, q = 0.f;
  for (int d = tid; d < DM; d += 256) { float v = hr[d]; s += v; q += v * v; }
  rs_[tid] = s; rq_[tid] = q; __syncthreads();
  for (int st = 128; st > 0; st >>= 1) {
    if (tid < st) { rs_[tid] += rs_[tid + st]; rq_[tid] += rq_[tid + st]; }
    __syncthreads();
  }
  float mu = rs_[0] * (1.0f / DM);
  float var = rq_[0] * (1.0f / DM) - mu * mu;
  float rinv = rsqrtf(var + 1e-5f);
  for (int d = tid; d < DM; d += 256) {
    float v = (hr[d] - mu) * rinv * gamma[d] + beta[d];
    outb[(size_t)row * DM + d] = (__bf16)(v * gr[d]);
  }
}

// W_re (DM,16) -> transposed bf16 (16,DM)
__global__ void k_wre_t(const float* __restrict__ wre, __bf16* __restrict__ wrb) {
  int i = blockIdx.x * blockDim.x + threadIdx.x;
  if (i < PATCHW * DM) {
    int p = i / DM, d = i % DM;
    wrb[i] = (__bf16)wre[(size_t)d * PATCHW + p];
  }
}

// final slice: out[b,t] = pre[b, (2048+t)/16, (2048+t)%16] + b_re
__global__ void k_gather(const float* __restrict__ pre, const float* __restrict__ bre,
                         float* __restrict__ out) {
  int t = blockIdx.x * blockDim.x + threadIdx.x;
  if (t < BB * PRED) {
    int b = t / PRED, j = t % PRED;
    int gidx = SEQ + j;
    int n = gidx >> 4, p = gidx & 15;
    out[t] = pre[((size_t)b * NPAT + n) * PATCHW + p] + bre[0];
  }
}

// ---------------------------------------------------------------------------
extern "C" void kernel_launch(void* const* d_in, const int* in_sizes, int n_in,
                              void* d_out, int out_size, void* d_ws, size_t ws_size,
                              hipStream_t stream) {
  (void)in_sizes; (void)n_in; (void)out_size; (void)ws_size;
  const float* x      = (const float*)d_in[0];
  const float* W_init = (const float*)d_in[1];
  const float* b_init = (const float*)d_in[2];
  const float* W_patch= (const float*)d_in[3];
  const float* b_patch= (const float*)d_in[4];
  const float* W_re   = (const float*)d_in[5];
  const float* b_re   = (const float*)d_in[6];
  const float* Wi = (const float*)d_in[7];  const float* bi = (const float*)d_in[8];
  const float* Wf = (const float*)d_in[9];  const float* bf = (const float*)d_in[10];
  const float* Wg = (const float*)d_in[11]; const float* bg = (const float*)d_in[12];
  const float* Wo = (const float*)d_in[13]; const float* bo = (const float*)d_in[14];
  const float* ln_g = (const float*)d_in[15];
  const float* ln_b = (const float*)d_in[16];
  float* out = (float*)d_out;
  char* ws = (char*)d_ws;

  size_t off = 0;
  auto alloc = [&](size_t bytes) { size_t o = off; off = (off + bytes + 255) & ~(size_t)255; return o; };
  __bf16* xb   = (__bf16*)(ws + alloc((size_t)BKR * SEQ * 2));
  __bf16* winb = (__bf16*)(ws + alloc((size_t)PRED * SEQ * 2));
  float*  trend= (float*) (ws + alloc((size_t)BKR * TROW * 4));
  __bf16* tpb  = (__bf16*)(ws + alloc((size_t)BN * 512 * 2));
  __bf16* wpb  = (__bf16*)(ws + alloc((size_t)DM * 512 * 2));
  float*  u    = (float*) (ws + alloc((size_t)BN * DM * 4));
  __bf16* ub   = (__bf16*)(ws + alloc((size_t)BN * DM * 2));
  __bf16* wib  = (__bf16*)(ws + alloc((size_t)DM * DM * 2));
  __bf16* wfb  = (__bf16*)(ws + alloc((size_t)DM * DM * 2));
  __bf16* wgb  = (__bf16*)(ws + alloc((size_t)DM * DM * 2));
  __bf16* wob  = (__bf16*)(ws + alloc((size_t)DM * DM * 2));
  float*  fbuf = (float*) (ws + alloc((size_t)BN * DM * 4));
  float*  gbuf = (float*) (ws + alloc((size_t)BN * DM * 4));
  float*  hbuf = (float*) (ws + alloc((size_t)BN * DM * 4));
  __bf16* wrb  = (__bf16*)(ws + alloc((size_t)PATCHW * DM * 2));
  float*  pre  = (float*) (ws + alloc((size_t)BN * PATCHW * 4));
  // aliased (reuse after last read, stream-ordered):
  float*  ibuf = u;               // u is dead after k_shiftgelu
  __bf16* hgb  = (__bf16*)trend;  // trend is dead after k_patchify (33.16MB each)
  __bf16* hob  = (__bf16*)fbuf;   // fbuf is dead after k_recur

  auto CEIL = [](size_t a, size_t b) { return (a + b - 1) / b; };
  const int T = 256;

  // 1) convert x, W_init to bf16; assemble trend buffer
  k_conv_bf16<<<CEIL((size_t)BKR * SEQ, T), T, 0, stream>>>(x, xb, (size_t)BKR * SEQ);
  k_conv_bf16<<<CEIL((size_t)PRED * SEQ, T), T, 0, stream>>>(W_init, winb, (size_t)PRED * SEQ);
  k_copy_x<<<CEIL((size_t)BKR * SEQ, T), T, 0, stream>>>(x, trend);

  // 2) trend_pred GEMM: (2048,2048) x (1990,2048)^T -> trend[:,2048:4038]
  gemm_wmma<<<dim3(BKR / 64, CEIL(PRED, 256)), T, 0, stream>>>(
      xb, SEQ, winb, SEQ, b_init, trend + SEQ, TROW, nullptr, 0,
      BKR, PRED, SEQ, 0);
  k_padfill<<<CEIL(BKR * 10, T), T, 0, stream>>>(trend);

  // 3) patch embedding GEMM: (16192,512) x (1024,512)^T -> u
  k_patchify<<<CEIL((size_t)BN * 512, T), T, 0, stream>>>(trend, tpb);
  k_conv_bf16<<<CEIL((size_t)DM * 512, T), T, 0, stream>>>(W_patch, wpb, (size_t)DM * 512);
  gemm_wmma<<<dim3(BN / 64, DM / 256), T, 0, stream>>>(
      tpb, 512, wpb, 512, b_patch, u, DM, nullptr, 0, BN, DM, 512, 0);

  // 4) shift + gelu -> ub (bf16)
  k_shiftgelu<<<CEIL((size_t)BN * DM, T), T, 0, stream>>>(u, ub);

  // 5) gate GEMMs (fused activations)
  k_conv_bf16<<<CEIL((size_t)DM * DM, T), T, 0, stream>>>(Wi, wib, (size_t)DM * DM);
  k_conv_bf16<<<CEIL((size_t)DM * DM, T), T, 0, stream>>>(Wf, wfb, (size_t)DM * DM);
  k_conv_bf16<<<CEIL((size_t)DM * DM, T), T, 0, stream>>>(Wg, wgb, (size_t)DM * DM);
  k_conv_bf16<<<CEIL((size_t)DM * DM, T), T, 0, stream>>>(Wo, wob, (size_t)DM * DM);
  gemm_wmma<<<dim3(BN / 64, DM / 256), T, 0, stream>>>(
      ub, DM, wib, DM, bi, ibuf, DM, nullptr, 0, BN, DM, DM, 1);   // silu
  gemm_wmma<<<dim3(BN / 64, DM / 256), T, 0, stream>>>(
      ub, DM, wfb, DM, bf, fbuf, DM, nullptr, 0, BN, DM, DM, 2);   // sigmoid
  gemm_wmma<<<dim3(BN / 64, DM / 256), T, 0, stream>>>(
      ub, DM, wgb, DM, bg, gbuf, DM, nullptr, 0, BN, DM, DM, 2);   // sigmoid

  // 6) recurrence h = f*h + (1-f)*i
  k_recur<<<CEIL(BB * DM, T), T, 0, stream>>>(fbuf, ibuf, hbuf);

  // 7) layernorm * gate -> bf16
  k_lngate<<<BN, T, 0, stream>>>(hbuf, gbuf, ln_g, ln_b, hgb);

  // 8) output projection GEMM -> bf16 directly (bias fused)
  gemm_wmma<<<dim3(BN / 64, DM / 256), T, 0, stream>>>(
      hgb, DM, wob, DM, bo, nullptr, 0, hob, DM, BN, DM, DM, 0);

  // 9) readout GEMM (N=16) and final slice
  k_wre_t<<<CEIL(PATCHW * DM, T), T, 0, stream>>>(W_re, wrb);
  gemm_wmma<<<dim3(BN / 64, 1), T, 0, stream>>>(
      hob, DM, wrb, DM, nullptr, pre, PATCHW, nullptr, 0, BN, PATCHW, DM, 0);
  k_gather<<<CEIL(BB * PRED, T), T, 0, stream>>>(pre, b_re, out);
}
